// PMCGNN_56693568307418
// MI455X (gfx1250) — matrix operand order repaired and checked by reading, not compiled, not run on hardware
//
#include <hip/hip_runtime.h>

// ---------------- problem constants ----------------
#define NN     10000
#define EE     120000
#define EIE    120000
#define ETOT   (EE + EIE)      // 240000
#define GG     100
#define FC     256
#define AIF    92
#define IBINS  64
#define NLAYER 3
#define MTILE  64              // edges per conv workgroup
#define LOG2F_ 0.69314718055994531f

typedef _Float16 h16;
typedef __attribute__((ext_vector_type(16))) _Float16 v16h;
typedef __attribute__((ext_vector_type(8)))  float    v8f;

// ---------------- WMMA helpers ----------------
// A fragment: 16x32 f16 tile, row-major.
// lane<16 : row=lane,    halves 0..7 = K k0..k0+7,  halves 8..15 = K k0+16..k0+23
// lane>=16: row=lane-16, halves 0..7 = K k0+8..+15, halves 8..15 = K k0+24..+31
__device__ __forceinline__ v16h load_a_frag(const h16* tile, int ld, int kt, int lane) {
    int row = lane & 15;
    int k0  = kt * 32 + ((lane & 16) ? 8 : 0);
    const h16* p = tile + row * ld + k0;
    v16h a;
    __builtin_memcpy(&a, p, 16);
    __builtin_memcpy(((char*)&a) + 16, p + 16, 16);
    return a;
}

// B fragment: packed so each lane reads 16 contiguous halves (32B).
__device__ __forceinline__ v16h load_b_frag(const h16* pk, int NT, int kt, int nt, int lane) {
    const h16* p = pk + ((((size_t)kt * NT + nt) * 32 + lane) << 4);
    v16h b;
    __builtin_memcpy(&b, p, 32);
    return b;
}

__device__ __forceinline__ v8f wmma_f16(v16h a, v16h b, v8f c) {
    return __builtin_amdgcn_wmma_f32_16x16x32_f16(false, a, false, b, (short)0, c, false, false);
}

// async 512B row copy: 32 lanes x b128, global -> LDS, tracked by ASYNCcnt
__device__ __forceinline__ void async_row_512(const h16* gsrc, h16* lds_dst, int lane) {
    unsigned     ldsa = (unsigned)(unsigned long long)lds_dst + (unsigned)(lane * 16);
    const char*  g    = (const char*)gsrc + lane * 16;
    asm volatile("global_load_async_to_lds_b128 %0, %1, off"
                 :: "v"(ldsa), "v"(g) : "memory");
}
__device__ __forceinline__ void wait_async0() {
    asm volatile("s_wait_asynccnt 0x0" ::: "memory");
}

__device__ __forceinline__ float silu_f(float x)  { return x / (1.f + expf(-x)); }
__device__ __forceinline__ float sigm_f(float x)  { return 1.f / (1.f + expf(-x)); }
__device__ __forceinline__ float sftp_f(float x)  { return fmaxf(x, 0.f) + log1pf(expf(-fabsf(x))); }

// ---------------- weight packing: f32 [K,Ncols] row-major -> WMMA B frags ----------------
__global__ void pack_b_kernel(const float* __restrict__ W, h16* __restrict__ out,
                              int K, int Ncols) {
    int idx = blockIdx.x * blockDim.x + threadIdx.x;
    if (idx >= K * Ncols) return;
    int j    = idx & 15;
    int lane = (idx >> 4) & 31;
    int rest = idx >> 9;
    int NT   = Ncols >> 4;
    int nt   = rest % NT;
    int kt   = rest / NT;
    int col  = nt * 16 + (lane & 15);
    int k    = kt * 32 + ((lane < 16) ? 0 : 8) + ((j < 8) ? j : (8 + j));
    out[idx] = (h16)W[(size_t)k * Ncols + col];
}

// ---------------- node embedding: node = x @ atom_W + atom_b ----------------
__global__ void node_embed_kernel(const float* __restrict__ x, const float* __restrict__ W,
                                  const float* __restrict__ b,
                                  float* __restrict__ n32, h16* __restrict__ n16) {
    int idx = blockIdx.x * blockDim.x + threadIdx.x;
    if (idx >= NN * FC) return;
    int n = idx >> 8, c = idx & 255;
    float acc = b[c];
    const float* xr = x + (size_t)n * AIF;
    for (int k = 0; k < AIF; ++k) acc += xr[k] * W[k * FC + c];
    n32[idx] = acc;
    n16[idx] = (h16)acc;
}

// ---------------- edge embedding: silu(rbf_gauss(-0.75/d) @ edge_W + b) ----------------
__global__ __launch_bounds__(32)
void edge_embed_kernel(const float* __restrict__ edge_attr, const h16* __restrict__ pkW,
                       const float* __restrict__ bias, h16* __restrict__ ef) {
    __shared__ __align__(16) h16 rbf[16 * FC];
    __shared__ float tval[16];
    int lane = threadIdx.x;
    int e0   = blockIdx.x * 16;
    if (lane < 16) tval[lane] = -0.75f / edge_attr[e0 + lane];
    __syncthreads();
    for (int i = 0; i < 128; ++i) {                 // 16*256 / 32 lanes
        int idx = i * 32 + lane;
        int row = idx >> 8, c = idx & 255;
        float center = -4.f + 8.f * (float)c * (1.f / 255.f);
        float base   = 31.875f * (tval[row] - center);
        rbf[idx] = (h16)expf(-base * base);
    }
    __syncthreads();
    for (int nt = 0; nt < 16; ++nt) {
        v8f acc = {};
        for (int kt = 0; kt < 8; ++kt)
            acc = wmma_f16(load_a_frag(rbf, FC, kt, lane), load_b_frag(pkW, 16, kt, nt, lane), acc);
        int col = nt * 16 + (lane & 15);
        float bc = bias[col];
        #pragma unroll
        for (int r = 0; r < 8; ++r) {
            int row = (lane < 16) ? r : (r + 8);
            ef[(size_t)(e0 + row) * FC + col] = (h16)silu_f(acc[r] + bc);
        }
    }
}

// ---------------- inf-edge embedding: bn(softplus(rbf_mq(pot) @ inf_W + b)) ----------------
__global__ __launch_bounds__(32)
void inf_embed_kernel(const float* __restrict__ inf_attr, const h16* __restrict__ pkW,
                      const float* __restrict__ bias, const float* __restrict__ bn,
                      h16* __restrict__ ef) {
    __shared__ __align__(16) h16 rbf[16 * IBINS];
    __shared__ float tval[16];
    int lane = threadIdx.x;
    int e0   = blockIdx.x * 16;
    if (lane < 16) {
        const float* a = inf_attr + (size_t)(e0 + lane) * 3;
        tval[lane] = a[0] * -0.801f + a[1] * -0.074f + a[2] * 0.145f;
    }
    __syncthreads();
    for (int i = 0; i < 32; ++i) {                  // 16*64 / 32 lanes
        int idx = i * 32 + lane;
        int row = idx >> 6, c = idx & 63;
        float center = -4.f + 8.f * (float)c * (1.f / 63.f);
        float base   = 7.875f * (tval[row] - center);
        rbf[idx] = (h16)sqrtf(1.f + base * base);
    }
    __syncthreads();
    for (int nt = 0; nt < 16; ++nt) {
        v8f acc = {};
        for (int kt = 0; kt < 2; ++kt)
            acc = wmma_f16(load_a_frag(rbf, IBINS, kt, lane), load_b_frag(pkW, 16, kt, nt, lane), acc);
        int col = nt * 16 + (lane & 15);
        float bc = bias[col];
        float g = bn[col], bb = bn[FC + col], mm = bn[2 * FC + col], vv = bn[3 * FC + col];
        float inv = rsqrtf(vv + 1e-5f) * g;
        #pragma unroll
        for (int r = 0; r < 8; ++r) {
            int row = (lane < 16) ? r : (r + 8);
            float v = (sftp_f(acc[r] + bc) - mm) * inv + bb;
            ef[(size_t)(e0 + row) * FC + col] = (h16)v;
        }
    }
}

// ---------------- conv layer: gated message passing, 64 edges / workgroup ----------------
__global__ __launch_bounds__(256)
void conv_edge_kernel(const int* __restrict__ edge_index, const int* __restrict__ inf_edge_index,
                      const h16* __restrict__ node16, const h16* __restrict__ ef,
                      const h16* __restrict__ pkW1f, const float* __restrict__ b1f,
                      const h16* __restrict__ pkW2f, const float* __restrict__ b2f,
                      const h16* __restrict__ pkW1,  const float* __restrict__ b1,
                      const h16* __restrict__ pkW2,  const float* __restrict__ b2,
                      const float* __restrict__ bni, float* __restrict__ agg) {
    __shared__ __align__(16) h16 htile[MTILE * 3 * FC];   // 96 KB
    __shared__ __align__(16) h16 z1f[MTILE * FC];         // 32 KB
    __shared__ __align__(16) h16 z1m[MTILE * FC];         // 32 KB
    __shared__ int sidx[MTILE], didx[MTILE];
    int tid  = threadIdx.x;
    int lane = tid & 31;
    int wave = tid >> 5;
    int e0   = blockIdx.x * MTILE;

    if (tid < MTILE) {
        int e = e0 + tid, s, d;
        if (e < EE) { s = edge_index[e];          d = edge_index[EE + e]; }
        else { int e2 = e - EE; s = inf_edge_index[e2]; d = inf_edge_index[EIE + e2]; }
        sidx[tid] = s; didx[tid] = d;
    }
    __syncthreads();

    // async gather h = [node[dst] | node[src] | ef] -> 64 x 768 f16
    // one global_load_async_to_lds_b128 (32 lanes x 16B = 512B) per (row, seg)
    for (int t = wave; t < MTILE * 3; t += 8) {
        int row = t / 3, seg = t % 3;
        const h16* srcp;
        if (seg == 0)      srcp = node16 + (size_t)didx[row] * FC;
        else if (seg == 1) srcp = node16 + (size_t)sidx[row] * FC;
        else               srcp = ef + (size_t)(e0 + row) * FC;
        async_row_512(srcp, &htile[row * 768 + seg * FC], lane);
    }
    wait_async0();
    __syncthreads();

    // stage 1: z1{f,m} = silu(h @ W1{f,m} + b1{f,m})
    // 32 (branch,ntile) tasks over 8 waves; each B frag reused for 4 row tiles
    for (int t = wave; t < 32; t += 8) {
        int br = t >> 4, nt = t & 15;
        const h16*  pk   = br ? pkW1 : pkW1f;
        const float* bia = br ? b1   : b1f;
        h16*        z    = br ? z1m  : z1f;
        v8f acc[4] = {};
        for (int kt = 0; kt < 24; ++kt) {
            v16h b = load_b_frag(pk, 16, kt, nt, lane);
            #pragma unroll
            for (int m = 0; m < 4; ++m)
                acc[m] = wmma_f16(load_a_frag(htile + m * 16 * 768, 768, kt, lane), b, acc[m]);
        }
        int col = nt * 16 + (lane & 15);
        float bc = bia[col];
        #pragma unroll
        for (int m = 0; m < 4; ++m) {
            #pragma unroll
            for (int r = 0; r < 8; ++r) {
                int row = m * 16 + ((lane < 16) ? r : (r + 8));
                z[row * FC + col] = (h16)silu_f(acc[m][r] + bc);
            }
        }
    }
    __syncthreads();

    // stage 2: zf = z1f@W2f+b2f ; zm = z1m@W2+b2 ; out = sigmoid(bn(zf)) * zm -> scatter
    for (int nt = wave; nt < 16; nt += 8) {
        v8f accf[4] = {}, accm[4] = {};
        for (int kt = 0; kt < 8; ++kt) {
            v16h bf = load_b_frag(pkW2f, 16, kt, nt, lane);
            v16h bm = load_b_frag(pkW2,  16, kt, nt, lane);
            #pragma unroll
            for (int m = 0; m < 4; ++m) {
                accf[m] = wmma_f16(load_a_frag(z1f + m * 16 * FC, FC, kt, lane), bf, accf[m]);
                accm[m] = wmma_f16(load_a_frag(z1m + m * 16 * FC, FC, kt, lane), bm, accm[m]);
            }
        }
        int col = nt * 16 + (lane & 15);
        float g = bni[col], bb = bni[FC + col], mm = bni[2 * FC + col], vv = bni[3 * FC + col];
        float inv = rsqrtf(vv + 1e-5f) * g;
        float bfc = b2f[col], bmc = b2[col];
        #pragma unroll
        for (int m = 0; m < 4; ++m) {
            #pragma unroll
            for (int r = 0; r < 8; ++r) {
                int row = m * 16 + ((lane < 16) ? r : (r + 8));
                float zfv = (accf[m][r] + bfc - mm) * inv + bb;
                float out = sigm_f(zfv) * (accm[m][r] + bmc);
                atomicAdd(&agg[(size_t)didx[row] * FC + col], out);
            }
        }
    }
}

// ---------------- node update: node = relu(node + bn(agg)); clear agg ----------------
__global__ void node_update_kernel(float* __restrict__ n32, h16* __restrict__ n16,
                                   float* __restrict__ agg, const float* __restrict__ bn) {
    int idx = blockIdx.x * blockDim.x + threadIdx.x;
    if (idx >= NN * FC) return;
    int col = idx & 255;
    float g = bn[col], bb = bn[FC + col], mm = bn[2 * FC + col], vv = bn[3 * FC + col];
    float a   = agg[idx];
    float val = n32[idx] + (a - mm) * rsqrtf(vv + 1e-5f) * g + bb;
    val = fmaxf(val, 0.f);
    n32[idx] = val;
    n16[idx] = (h16)val;
    agg[idx] = 0.f;
}

// ---------------- pooling ----------------
__global__ void cnt_kernel(const int* __restrict__ batch, float* __restrict__ cnt) {
    int n = blockIdx.x * blockDim.x + threadIdx.x;
    if (n < NN) atomicAdd(&cnt[batch[n]], 1.f);
}
__global__ void pool_kernel(const float* __restrict__ n32, const int* __restrict__ batch,
                            float* __restrict__ pooled) {
    int idx = blockIdx.x * blockDim.x + threadIdx.x;
    if (idx >= NN * FC) return;
    int n = idx >> 8, c = idx & 255;
    atomicAdd(&pooled[(size_t)batch[n] * FC + c], n32[idx]);
}

// ---------------- head ----------------
__global__ __launch_bounds__(256)
void head_kernel(const float* __restrict__ pooled, const float* __restrict__ cnt,
                 const float* __restrict__ fcW, const float* __restrict__ fcb,
                 const float* __restrict__ outW, const float* __restrict__ outb,
                 float* __restrict__ out) {
    __shared__ float mean[FC];
    __shared__ float red[FC];
    int g = blockIdx.x, t = threadIdx.x;
    float c = fmaxf(cnt[g], 1.f);
    mean[t] = pooled[(size_t)g * FC + t] / c;
    __syncthreads();
    float acc = fcb[t];
    for (int k = 0; k < FC; ++k) acc += mean[k] * fcW[k * FC + t];
    red[t] = (sftp_f(acc) - LOG2F_) * outW[t];
    __syncthreads();
    for (int s = 128; s > 0; s >>= 1) {
        if (t < s) red[t] += red[t + s];
        __syncthreads();
    }
    if (t == 0) out[g] = red[0] + outb[0];
}

// ---------------- host launcher ----------------
extern "C" void kernel_launch(void* const* d_in, const int* in_sizes, int n_in,
                              void* d_out, int out_size, void* d_ws, size_t ws_size,
                              hipStream_t stream) {
    const float* x        = (const float*)d_in[0];
    const int*   ei       = (const int*)  d_in[1];
    const float* ea       = (const float*)d_in[2];
    const int*   iei      = (const int*)  d_in[3];
    const float* iea      = (const float*)d_in[4];
    const int*   batch    = (const int*)  d_in[5];
    const float* atom_W   = (const float*)d_in[6];
    const float* atom_b   = (const float*)d_in[7];
    const float* edge_W   = (const float*)d_in[8];
    const float* edge_b   = (const float*)d_in[9];
    const float* inf_W    = (const float*)d_in[10];
    const float* inf_b    = (const float*)d_in[11];
    const float* inf_bn   = (const float*)d_in[12];
    const float* conv_W1f = (const float*)d_in[13];
    const float* conv_b1f = (const float*)d_in[14];
    const float* conv_W2f = (const float*)d_in[15];
    const float* conv_b2f = (const float*)d_in[16];
    const float* conv_W1  = (const float*)d_in[17];
    const float* conv_b1  = (const float*)d_in[18];
    const float* conv_W2  = (const float*)d_in[19];
    const float* conv_b2  = (const float*)d_in[20];
    const float* conv_bn  = (const float*)d_in[21];
    const float* conv_bni = (const float*)d_in[22];
    const float* fc_W     = (const float*)d_in[23];
    const float* fc_b     = (const float*)d_in[24];
    const float* out_W    = (const float*)d_in[25];
    const float* out_b    = (const float*)d_in[26];
    float* out = (float*)d_out;

    // workspace carve-up (256B aligned)
    size_t off = 0;
    char* base = (char*)d_ws;
    auto carve = [&](size_t bytes) -> void* {
        void* p = base + off;
        off += (bytes + 255) & ~(size_t)255;
        return p;
    };
    h16*   ef      = (h16*)  carve((size_t)ETOT * FC * 2);
    float* node32  = (float*)carve((size_t)NN * FC * 4);
    h16*   node16  = (h16*)  carve((size_t)NN * FC * 2);
    float* agg     = (float*)carve((size_t)NN * FC * 4);
    float* pooled  = (float*)carve((size_t)GG * FC * 4);
    float* cnt     = (float*)carve((size_t)GG * 4);
    h16*   pk_edge = (h16*)  carve((size_t)FC * FC * 2);
    h16*   pk_inf  = (h16*)  carve((size_t)IBINS * FC * 2);
    h16*   pk_W1f  = (h16*)  carve((size_t)NLAYER * 3 * FC * FC * 2);
    h16*   pk_W1   = (h16*)  carve((size_t)NLAYER * 3 * FC * FC * 2);
    h16*   pk_W2f  = (h16*)  carve((size_t)NLAYER * FC * FC * 2);
    h16*   pk_W2   = (h16*)  carve((size_t)NLAYER * FC * FC * 2);

    hipMemsetAsync(agg,    0, (size_t)NN * FC * 4, stream);
    hipMemsetAsync(pooled, 0, (size_t)GG * FC * 4, stream);
    hipMemsetAsync(cnt,    0, (size_t)GG * 4,      stream);

    auto pack = [&](const float* W, h16* dst, int K, int Ncols) {
        int tot = K * Ncols;
        pack_b_kernel<<<(tot + 255) / 256, 256, 0, stream>>>(W, dst, K, Ncols);
    };
    pack(edge_W, pk_edge, FC, FC);
    pack(inf_W,  pk_inf,  IBINS, FC);
    for (int l = 0; l < NLAYER; ++l) {
        pack(conv_W1f + (size_t)l * 3 * FC * FC, pk_W1f + (size_t)l * 3 * FC * FC, 3 * FC, FC);
        pack(conv_W1  + (size_t)l * 3 * FC * FC, pk_W1  + (size_t)l * 3 * FC * FC, 3 * FC, FC);
        pack(conv_W2f + (size_t)l * FC * FC,     pk_W2f + (size_t)l * FC * FC,     FC, FC);
        pack(conv_W2  + (size_t)l * FC * FC,     pk_W2  + (size_t)l * FC * FC,     FC, FC);
    }

    node_embed_kernel<<<(NN * FC + 255) / 256, 256, 0, stream>>>(x, atom_W, atom_b, node32, node16);
    edge_embed_kernel<<<EE / 16, 32, 0, stream>>>(ea, pk_edge, edge_b, ef);
    inf_embed_kernel<<<EIE / 16, 32, 0, stream>>>(iea, pk_inf, inf_b, inf_bn,
                                                  ef + (size_t)EE * FC);

    for (int l = 0; l < NLAYER; ++l) {
        conv_edge_kernel<<<ETOT / MTILE, 256, 0, stream>>>(
            ei, iei, node16, ef,
            pk_W1f + (size_t)l * 3 * FC * FC, conv_b1f + (size_t)l * FC,
            pk_W2f + (size_t)l * FC * FC,     conv_b2f + (size_t)l * FC,
            pk_W1  + (size_t)l * 3 * FC * FC, conv_b1  + (size_t)l * FC,
            pk_W2  + (size_t)l * FC * FC,     conv_b2  + (size_t)l * FC,
            conv_bni + (size_t)l * 4 * FC, agg);
        node_update_kernel<<<(NN * FC + 255) / 256, 256, 0, stream>>>(
            node32, node16, agg, conv_bn + (size_t)l * 4 * FC);
    }

    cnt_kernel<<<(NN + 255) / 256, 256, 0, stream>>>(batch, cnt);
    pool_kernel<<<(NN * FC + 255) / 256, 256, 0, stream>>>(node32, batch, pooled);
    head_kernel<<<GG, 256, 0, stream>>>(pooled, cnt, fc_W, fc_b, out_W, out_b, out);

    (void)in_sizes; (void)n_in; (void)out_size; (void)ws_size;
}
